// CasualSelfAttention_53206054863346
// MI455X (gfx1250) — matrix operand order repaired and checked by reading, not compiled, not run on hardware
//
#include <hip/hip_runtime.h>
#include <math.h>

// ---------------------------------------------------------------------------
// Causal self-attention forward for gfx1250 (MI455X), bf16 WMMA / fp32 accum.
// B=2, T=2048, C=1024, H=16, D=64.
// Pipelined: operand double-buffering so loads overlap v_wmma issue.
// ---------------------------------------------------------------------------

typedef __bf16 bf16;
typedef __attribute__((ext_vector_type(16))) __bf16 v16bf;
typedef __attribute__((ext_vector_type(8)))  __bf16 v8bf;
typedef __attribute__((ext_vector_type(8)))  float  v8f;

#define B_  2
#define T_  2048
#define C_  1024
#define H_  16
#define D_  64
#define M_  (B_ * T_)      // 4096 rows
#define N1_ (3 * C_)       // 3072 qkv columns

__device__ __forceinline__ v8f wmma_bf16(v16bf a, v16bf b, v8f c) {
  // D(f32 16x16) = A(bf16 16x32) * B(bf16 32x16) + C
  return __builtin_amdgcn_wmma_f32_16x16x32_bf16(
      /*neg_a=*/false, a, /*neg_b=*/false, b,
      /*c_mod=*/(short)0, c, /*reuse_a=*/false, /*reuse_b=*/false);
}

__device__ __forceinline__ v8f vzero8() {
  v8f z;
#pragma unroll
  for (int i = 0; i < 8; ++i) z[i] = 0.0f;
  return z;
}

__device__ __forceinline__ v16bf join8(v8bf lo, v8bf hi) {
  v16bf r;
#pragma unroll
  for (int i = 0; i < 8; ++i) { r[i] = lo[i]; r[8 + i] = hi[i]; }
  return r;
}

// A-operand (16x32 bf16) from a row-major [*, stride] bf16 matrix.
// lane(lo,hi): row = row0+lo; elems e -> k = kk+16*hi+e
__device__ __forceinline__ v16bf loadA(const bf16* base, int row, int stride,
                                       int kk, int hi) {
  const bf16* p = base + (size_t)row * stride + kk + hi * 16;
  return *(const v16bf*)p;
}

// B-operand (32x16 bf16) from a row-major [*, stride] bf16 matrix holding B^T
// (so a B column is a contiguous row). lane(lo,hi): col = lo;
// elems e -> k = kk + 16*hi + e  (contiguous 32B load)
__device__ __forceinline__ v16bf loadB(const bf16* baseT, int col, int stride,
                                       int kk, int hi) {
  return *(const v16bf*)(baseT + (size_t)col * stride + kk + hi * 16);
}

// ---------------------------------------------------------------------------
// Stage 0: fp32 -> bf16 pack / transpose
// ---------------------------------------------------------------------------
__global__ void cvt_bf16_kernel(const float* __restrict__ src,
                                bf16* __restrict__ dst, int n) {
  int i = blockIdx.x * blockDim.x + threadIdx.x;
  if (i < n) dst[i] = (bf16)src[i];
}

// src: rows x cols (row-major) -> dst: cols x rows (row-major), bf16
__global__ void transpose_bf16_kernel(const float* __restrict__ src,
                                      bf16* __restrict__ dst,
                                      int rows, int cols, int n) {
  int i = blockIdx.x * blockDim.x + threadIdx.x;
  if (i < n) {
    int r = i / rows;   // dst row == src col
    int k = i % rows;   // dst col == src row
    dst[i] = (bf16)src[(size_t)k * cols + r];
  }
}

// ---------------------------------------------------------------------------
// Stage 1: fused QKV GEMM  (M_ x C_) @ (C_ x N1_) + bias, pipelined.
// One wave computes a 16x64 tile (4 WMMA accumulators sharing A).
// ---------------------------------------------------------------------------
__global__ void qkv_gemm_kernel(const bf16* __restrict__ xb,
                                const bf16* __restrict__ waT,
                                const float* __restrict__ b_attn,
                                bf16* __restrict__ qb, bf16* __restrict__ kb,
                                bf16* __restrict__ vbT) {
  const int lane = threadIdx.x & 31;
  const int wave = (blockIdx.x * blockDim.x + threadIdx.x) >> 5;
  const int tiles_n = N1_ / 64;                 // 48
  const int tm = wave / tiles_n;
  const int tn = wave % tiles_n;
  if (tm >= M_ / 16) return;                    // wave-uniform
  const int hi = lane >> 4;
  const int lo = lane & 15;
  const int arow = tm * 16 + lo;

  v8f acc[4];
#pragma unroll
  for (int nn = 0; nn < 4; ++nn) acc[nn] = vzero8();

  // ---- software pipeline: prologue loads for kk = 0 ----
  v16bf aCur = loadA(xb, arow, C_, 0, hi);
  v16bf bCur[4];
#pragma unroll
  for (int nn = 0; nn < 4; ++nn)
    bCur[nn] = loadB(waT, tn * 64 + nn * 16 + lo, C_, 0, hi);

  for (int kk = 32; kk < C_; kk += 32) {
    // issue next-step loads BEFORE consuming current operands
    v16bf aNext = loadA(xb, arow, C_, kk, hi);
    v16bf bNext[4];
#pragma unroll
    for (int nn = 0; nn < 4; ++nn)
      bNext[nn] = loadB(waT, tn * 64 + nn * 16 + lo, C_, kk, hi);
#pragma unroll
    for (int nn = 0; nn < 4; ++nn)
      acc[nn] = wmma_bf16(aCur, bCur[nn], acc[nn]);
    aCur = aNext;
#pragma unroll
    for (int nn = 0; nn < 4; ++nn) bCur[nn] = bNext[nn];
  }
#pragma unroll
  for (int nn = 0; nn < 4; ++nn)
    acc[nn] = wmma_bf16(aCur, bCur[nn], acc[nn]);

  // ---- epilogue: bias, scale q, scatter to (B,H,T,D) / (B,H,D,T) ----
#pragma unroll
  for (int nn = 0; nn < 4; ++nn) {
    const int j = tn * 64 + nn * 16 + lo;       // output column in [0,3072)
    const float bias = b_attn[j];
    const int part = j >> 10;                   // 0=q 1=k 2=v
    const int c    = j & (C_ - 1);
    const int head = c >> 6;
    const int d    = c & (D_ - 1);
#pragma unroll
    for (int v = 0; v < 8; ++v) {
      const int i   = tm * 16 + v + 8 * hi;     // output row
      const int bb  = i >> 11;                  // batch
      const int t   = i & (T_ - 1);
      const float val = acc[nn][v] + bias;
      const size_t bh = (size_t)(bb * H_ + head);
      if (part == 0)      qb [(bh * T_ + t) * D_ + d] = (bf16)(val * 0.125f);
      else if (part == 1) kb [(bh * T_ + t) * D_ + d] = (bf16)val;
      else                vbT[(bh * D_ + d) * T_ + t] = (bf16)val;
    }
  }
}

// ---------------------------------------------------------------------------
// Stage 2: flash attention. One wave per (b,h, 16-query-row block).
// Computes S^T = K * Q^T so the online softmax reduces per-lane, and the
// two 16-key S^T tiles repack into the 16x32 A-operand of P with NO
// cross-lane data movement. K and V tiles for the step are issued up front
// so the V waits hide behind the softmax VALU work.
// ---------------------------------------------------------------------------
__global__ void attn_kernel(const bf16* __restrict__ qb,
                            const bf16* __restrict__ kb,
                            const bf16* __restrict__ vbT,
                            bf16* __restrict__ ya) {
  const int lane = threadIdx.x & 31;
  const int wave = (blockIdx.x * blockDim.x + threadIdx.x) >> 5;
  const int rowblocks = T_ / 16;                // 128
  const int bh   = wave / rowblocks;
  const int qblk = wave % rowblocks;
  if (bh >= B_ * H_) return;                    // wave-uniform
  const int q0 = qblk * 16;
  const int hi = lane >> 4;
  const int lo = lane & 15;

  const bf16* Q = qb  + (size_t)bh * T_ * D_;
  const bf16* K = kb  + (size_t)bh * T_ * D_;
  const bf16* V = vbT + (size_t)bh * D_ * T_;

  // Q^T B-operand (loop invariant): b[e] = Q[q0+lo][kk*32 + hi*16 + e]
  v16bf bq[2];
#pragma unroll
  for (int kk = 0; kk < 2; ++kk)
    bq[kk] = *(const v16bf*)(Q + (size_t)(q0 + lo) * D_ + kk * 32 + hi * 16);

  v8f o[4];
#pragma unroll
  for (int d0 = 0; d0 < 4; ++d0) o[d0] = vzero8();
  float m_run = -1e30f, l_run = 0.0f;

  for (int j = 0; j < q0 + 16; j += 32) {       // 32 keys per step
    // ---- issue ALL loads for this step up front ----
    v16bf kA[2][2];                             // [key-tile][kk]
#pragma unroll
    for (int kk = 0; kk < 2; ++kk) {
      kA[0][kk] = loadA(K, j + lo,      D_, kk * 32, hi);
      kA[1][kk] = loadA(K, j + 16 + lo, D_, kk * 32, hi);
    }
    v16bf bv[4];
#pragma unroll
    for (int d0 = 0; d0 < 4; ++d0)
      bv[d0] = join8(*(const v8bf*)(V + (size_t)(d0 * 16 + lo) * T_ + j + hi * 8),
                     *(const v8bf*)(V + (size_t)(d0 * 16 + lo) * T_ + j + 16 + hi * 8));

    // speculative prefetch of next step's K/V lines (global_prefetch_b8)
    __builtin_prefetch(K + (size_t)(j + 32 + lo) * D_, 0, 3);
    __builtin_prefetch(V + (size_t)lo * T_ + j + 32, 0, 3);

    // ---- S^T = K * Q^T ----
    v8f st0 = vzero8(), st1 = vzero8();
#pragma unroll
    for (int kk = 0; kk < 2; ++kk) {
      st0 = wmma_bf16(kA[0][kk], bq[kk], st0);
      st1 = wmma_bf16(kA[1][kk], bq[kk], st1);
    }

    // S^T C/D layout: lane holds column t=q0+lo, rows s = j + v + 8*hi (+16)
    const int tg = q0 + lo;
    float p0[8], p1[8];
    float bm = -1e30f;
#pragma unroll
    for (int v = 0; v < 8; ++v) {
      const int s0 = j + v + 8 * hi;
      p0[v] = (s0      <= tg) ? st0[v] : -1e30f;   // causal mask
      p1[v] = (s0 + 16 <= tg) ? st1[v] : -1e30f;
      bm = fmaxf(bm, fmaxf(p0[v], p1[v]));
    }
    bm = fmaxf(bm, __shfl_xor(bm, 16, 32));     // combine the two s-halves
    const float m_new = fmaxf(m_run, bm);
    const float scale_prev = __expf(m_run - m_new);

    float bs = 0.0f;
    v16bf pa;                                    // P as 16x32 A-operand
#pragma unroll
    for (int v = 0; v < 8; ++v) {
      const float e0 = __expf(p0[v] - m_new);
      const float e1 = __expf(p1[v] - m_new);
      bs += e0 + e1;
      pa[v]     = (bf16)e0;                      // elems 0..7: tile0 s-rows
      pa[8 + v] = (bf16)e1;                      // elems 8..15: tile1 s-rows
    }
    bs += __shfl_xor(bs, 16, 32);
    l_run = l_run * scale_prev + bs;
    m_run = m_new;

    // Rescale O (rows t = v + 8*hi; factor lives in lane t)
#pragma unroll
    for (int v = 0; v < 8; ++v) {
      const float f = __shfl(scale_prev, v + 8 * hi, 32);
#pragma unroll
      for (int d0 = 0; d0 < 4; ++d0) o[d0][v] *= f;
    }

    // O += P(16x32) x V(32x16) for each 16-wide d block
#pragma unroll
    for (int d0 = 0; d0 < 4; ++d0)
      o[d0] = wmma_bf16(pa, bv[d0], o[d0]);
  }

  // Normalize rows and write (B,T,C) bf16 for the output projection.
  const float inv = 1.0f / l_run;
  const int bb = bh / H_, h = bh % H_;
#pragma unroll
  for (int v = 0; v < 8; ++v) {
    const float f = __shfl(inv, v + 8 * hi, 32);
    const int tg = q0 + v + 8 * hi;
#pragma unroll
    for (int d0 = 0; d0 < 4; ++d0) {
      ya[((size_t)(bb * T_ + tg)) * C_ + h * D_ + d0 * 16 + lo] =
          (bf16)(o[d0][v] * f);
    }
  }
}

// ---------------------------------------------------------------------------
// Stage 3: output projection (M_ x C_) @ (C_ x C_) + bias -> fp32 out,
// pipelined like stage 1.
// ---------------------------------------------------------------------------
__global__ void proj_gemm_kernel(const bf16* __restrict__ ya,
                                 const bf16* __restrict__ wpT,
                                 const float* __restrict__ b_proj,
                                 float* __restrict__ out) {
  const int lane = threadIdx.x & 31;
  const int wave = (blockIdx.x * blockDim.x + threadIdx.x) >> 5;
  const int tiles_n = C_ / 64;                  // 16
  const int tm = wave / tiles_n;
  const int tn = wave % tiles_n;
  if (tm >= M_ / 16) return;                    // wave-uniform
  const int hi = lane >> 4;
  const int lo = lane & 15;
  const int arow = tm * 16 + lo;

  v8f acc[4];
#pragma unroll
  for (int nn = 0; nn < 4; ++nn) acc[nn] = vzero8();

  v16bf aCur = loadA(ya, arow, C_, 0, hi);
  v16bf bCur[4];
#pragma unroll
  for (int nn = 0; nn < 4; ++nn)
    bCur[nn] = loadB(wpT, tn * 64 + nn * 16 + lo, C_, 0, hi);

  for (int kk = 32; kk < C_; kk += 32) {
    v16bf aNext = loadA(ya, arow, C_, kk, hi);
    v16bf bNext[4];
#pragma unroll
    for (int nn = 0; nn < 4; ++nn)
      bNext[nn] = loadB(wpT, tn * 64 + nn * 16 + lo, C_, kk, hi);
#pragma unroll
    for (int nn = 0; nn < 4; ++nn)
      acc[nn] = wmma_bf16(aCur, bCur[nn], acc[nn]);
    aCur = aNext;
#pragma unroll
    for (int nn = 0; nn < 4; ++nn) bCur[nn] = bNext[nn];
  }
#pragma unroll
  for (int nn = 0; nn < 4; ++nn)
    acc[nn] = wmma_bf16(aCur, bCur[nn], acc[nn]);

#pragma unroll
  for (int nn = 0; nn < 4; ++nn) {
    const int j = tn * 64 + nn * 16 + lo;
    const float bias = b_proj[j];
#pragma unroll
    for (int v = 0; v < 8; ++v) {
      const int i = tm * 16 + v + 8 * hi;
      out[(size_t)i * C_ + j] = acc[nn][v] + bias;
    }
  }
}

// ---------------------------------------------------------------------------
// Launch
// ---------------------------------------------------------------------------
extern "C" void kernel_launch(void* const* d_in, const int* in_sizes, int n_in,
                              void* d_out, int out_size, void* d_ws,
                              size_t ws_size, hipStream_t stream) {
  const float* x      = (const float*)d_in[0];
  const float* w_attn = (const float*)d_in[1];
  const float* b_attn = (const float*)d_in[2];
  const float* w_proj = (const float*)d_in[3];
  const float* b_proj = (const float*)d_in[4];
  float* out = (float*)d_out;

  // Workspace layout (48 MiB total, all offsets 2 MiB-aligned):
  bf16* xb  = (bf16*)d_ws;                      // (M_,C_)     x in bf16
  bf16* waT = xb  + (size_t)M_  * C_;           // (N1_,C_)    w_attn^T
  bf16* wpT = waT + (size_t)N1_ * C_;           // (C_,C_)     w_proj^T
  bf16* qb  = wpT + (size_t)C_  * C_;           // (B,H,T,D)   q * 1/sqrt(D)
  bf16* kb  = qb  + (size_t)B_ * H_ * T_ * D_;  // (B,H,T,D)   k
  bf16* vbT = kb  + (size_t)B_ * H_ * T_ * D_;  // (B,H,D,T)   v transposed
  bf16* ya  = vbT + (size_t)B_ * H_ * T_ * D_;  // (B,T,C)     attn output

  { int n = M_ * C_;
    cvt_bf16_kernel<<<(n + 255) / 256, 256, 0, stream>>>(x, xb, n); }
  { int n = N1_ * C_;
    transpose_bf16_kernel<<<(n + 255) / 256, 256, 0, stream>>>(w_attn, waT, C_, N1_, n); }
  { int n = C_ * C_;
    transpose_bf16_kernel<<<(n + 255) / 256, 256, 0, stream>>>(w_proj, wpT, C_, C_, n); }

  { int waves = (M_ / 16) * (N1_ / 64);         // 12288 waves
    qkv_gemm_kernel<<<(waves * 32 + 255) / 256, 256, 0, stream>>>(
        xb, waT, b_attn, qb, kb, vbT); }
  { int waves = B_ * H_ * (T_ / 16);            // 4096 waves
    attn_kernel<<<(waves * 32 + 255) / 256, 256, 0, stream>>>(qb, kb, vbT, ya); }
  { int waves = (M_ / 16) * (C_ / 64);          // 4096 waves
    proj_gemm_kernel<<<(waves * 32 + 255) / 256, 256, 0, stream>>>(
        ya, wpT, b_proj, out); }
}
